// GatedGCNNet_3753801417624
// MI455X (gfx1250) — compile-verified, matrix operand order checked by reading, not run on hardware
//
#include <hip/hip_runtime.h>
#include <hip/hip_bf16.h>

#define HID   70
#define HP    96          // padded hidden (3 x K32 chunks, 6 x N16 tiles)
#define NLAY  4
#define NCLS  10

typedef __attribute__((ext_vector_type(16))) __bf16 v16bf;
typedef __attribute__((ext_vector_type(8)))  float  v8f;

// ---------- helpers ----------
__device__ inline unsigned short f2bf_bits(float f) {
    union { float f; unsigned u; } v; v.f = f;
    return (unsigned short)((v.u + 0x7FFFu + ((v.u >> 16) & 1u)) >> 16);   // RNE
}
__device__ inline __bf16 bits2bf(unsigned short s) {
    __bf16 b; __builtin_memcpy(&b, &s, 2); return b;
}

// ---------- weight packing: f32 (K0 x N0, row-major) -> bf16 B-fragment layout ----------
// out layout: [K/32 chunk][N/16 tile][lane 0..31][elem 0..15]  (contiguous 32B per lane)
// lane -> N = tile*16 + (lane&15);  elem e -> K = kc*32 + ((lane>=16)?8:0) + (e<8 ? e : 16+(e-8))
__global__ void pack_weight(const float* __restrict__ W, unsigned short* __restrict__ out,
                            int K0, int N0, int KP, int NP) {
    int total = KP * NP;
    for (int idx = blockIdx.x * blockDim.x + threadIdx.x; idx < total;
         idx += gridDim.x * blockDim.x) {
        int e    = idx & 15;
        int lane = (idx >> 4) & 31;
        int chk  = idx >> 9;
        int nCT  = NP >> 4;
        int ct   = chk % nCT;
        int kc   = chk / nCT;
        int n    = ct * 16 + (lane & 15);
        int k    = kc * 32 + ((lane >> 4) << 3) + ((e < 8) ? e : 16 + (e - 8));
        float v  = (k < K0 && n < N0) ? W[(size_t)k * N0 + n] : 0.0f;
        out[idx] = f2bf_bits(v);
    }
}

// ---------- WMMA GEMM:  C[M x NP] = A[M x K] @ Wpacked + bias, optional ReLU ----------
// One wave owns a 16-row strip: A fragments for all K-chunks kept in VGPRs (read once),
// loops over all NCT column tiles with per-tile v8f accumulators (fully unrolled).
// A: f32 row-major, row stride == K.  K = NKC*32, NP = NCT*16.
template <int NKC, int NCT>
__global__ void gemm_wmma(const float* __restrict__ A, const unsigned short* __restrict__ Bp,
                          const float* __restrict__ bias, float* __restrict__ C,
                          int M, int N0, int relu) {
    constexpr int K  = NKC * 32;
    constexpr int NP = NCT * 16;

    int rt   = blockIdx.x * (blockDim.x >> 5) + (threadIdx.x >> 5);
    int lane = threadIdx.x & 31;
    int nRT  = (M + 15) >> 4;
    if (rt >= nRT) return;                       // whole-wave exit: EXEC stays all-ones

    int row  = rt * 16 + (lane & 15);
    int rowc = row < M ? row : (M - 1);          // clamp for tail-tile loads
    int koff = (lane >> 4) << 3;                 // 0 | 8 (per ISA A-fragment layout)

    // Load the whole 16 x K activation strip once (per-lane: two 32B runs per chunk).
    v16bf afrag[NKC];
#pragma unroll
    for (int kc = 0; kc < NKC; ++kc) {
        const float* Ar = A + (size_t)rowc * K + kc * 32 + koff;
#pragma unroll
        for (int i = 0; i < 8; ++i) {
            afrag[kc][i]     = bits2bf(f2bf_bits(Ar[i]));
            afrag[kc][i + 8] = bits2bf(f2bf_bits(Ar[i + 16]));
        }
    }

    v8f acc[NCT];
#pragma unroll
    for (int ct = 0; ct < NCT; ++ct) {
        acc[ct] = {};
#pragma unroll
        for (int kc = 0; kc < NKC; ++kc) {
            const v16bf* bp =
                (const v16bf*)(Bp + (((size_t)(kc * NCT + ct) * 32 + lane) << 4));
            v16bf b = *bp;
            acc[ct] = __builtin_amdgcn_wmma_f32_16x16x32_bf16(false, afrag[kc], false, b,
                                                              (short)0, acc[ct],
                                                              false, false);
        }
    }

    int rbase = rt * 16 + ((lane >> 4) << 3);
#pragma unroll
    for (int ct = 0; ct < NCT; ++ct) {
        int ncol = ct * 16 + (lane & 15);
        float bv = (ncol < N0) ? bias[ncol] : 0.0f;
#pragma unroll
        for (int r = 0; r < 8; ++r) {
            int mr = rbase + r;
            if (mr < M) {
                float v = acc[ct][r] + bv;
                if (relu) v = fmaxf(v, 0.0f);
                C[(size_t)mr * NP + ncol] = v;
            }
        }
    }
}

// ---------- edge stage: e_hat = Ce + Dh[src] + Eh[dst]; sigma; segment sums ----------
__global__ void edge_kernel(float* __restrict__ ehat /* in: Ce, overwritten */,
                            const float* __restrict__ Dh, const float* __restrict__ Eh,
                            const float* __restrict__ Bh,
                            const int* __restrict__ src, const int* __restrict__ dst,
                            float* __restrict__ num, float* __restrict__ den, int E) {
    long total = (long)E * HID;
    long idx   = (long)blockIdx.x * blockDim.x + threadIdx.x;
    if (idx >= total) return;
    int ed = (int)(idx / HID);
    int c  = (int)(idx - (long)ed * HID);
    int s  = src[ed], d = dst[ed];
    float eh  = ehat[(size_t)ed * HP + c] + Dh[(size_t)s * HP + c] + Eh[(size_t)d * HP + c];
    float sig = 1.0f / (1.0f + __expf(-eh));
    ehat[(size_t)ed * HP + c] = eh;
    atomicAdd(&num[(size_t)d * HP + c], sig * Bh[(size_t)s * HP + c]);
    atomicAdd(&den[(size_t)d * HP + c], sig);
}

// ---------- h_pre = Ah + num / (den + 1e-6)  (in place over Ah) ----------
__global__ void combine_h(float* __restrict__ hpre, const float* __restrict__ num,
                          const float* __restrict__ den, int N) {
    long total = (long)N * HID;
    long idx   = (long)blockIdx.x * blockDim.x + threadIdx.x;
    if (idx >= total) return;
    int n = (int)(idx / HID);
    int c = (int)(idx - (long)n * HID);
    size_t o = (size_t)n * HP + c;
    hpre[o] = hpre[o] + num[o] / (den[o] + 1e-6f);
}

// ---------- per-column sums / sumsq of (buf * rowscale) via LDS atomics ----------
__global__ void col_stats(const float* __restrict__ buf, const float* __restrict__ rowscale,
                          int M, float* __restrict__ sums /* [2*HP] */) {
    __shared__ float s_sum[HID];
    __shared__ float s_sq[HID];
    for (int i = threadIdx.x; i < HID; i += blockDim.x) { s_sum[i] = 0.f; s_sq[i] = 0.f; }
    __syncthreads();
    long total  = (long)M * HID;
    long stride = (long)gridDim.x * blockDim.x;
    for (long idx = (long)blockIdx.x * blockDim.x + threadIdx.x; idx < total; idx += stride) {
        int n = (int)(idx / HID);
        int c = (int)(idx - (long)n * HID);
        float x = buf[(size_t)n * HP + c] * rowscale[n];
        atomicAdd(&s_sum[c], x);
        atomicAdd(&s_sq[c], x * x);
    }
    __syncthreads();
    for (int i = threadIdx.x; i < HID; i += blockDim.x) {
        atomicAdd(&sums[i],      s_sum[i]);
        atomicAdd(&sums[HP + i], s_sq[i]);
    }
}

__global__ void finalize_stats(const float* __restrict__ sums, float* __restrict__ meaninv,
                               float invM) {
    int c = threadIdx.x;
    if (c < HID) {
        float m = sums[c] * invM;
        float v = sums[HP + c] * invM - m * m;
        meaninv[c]      = m;
        meaninv[HP + c] = rsqrtf(v + 1e-5f);
    }
}

// ---------- io += relu( bn(pre * rowscale) * g + b ) ----------
__global__ void bn_residual(float* __restrict__ io, const float* __restrict__ pre,
                            const float* __restrict__ rowscale,
                            const float* __restrict__ meaninv,
                            const float* __restrict__ g, const float* __restrict__ b, int M) {
    long total = (long)M * HID;
    long idx   = (long)blockIdx.x * blockDim.x + threadIdx.x;
    if (idx >= total) return;
    int n = (int)(idx / HID);
    int c = (int)(idx - (long)n * HID);
    float x = pre[(size_t)n * HP + c] * rowscale[n];
    float y = (x - meaninv[c]) * meaninv[HP + c] * g[c] + b[c];
    io[(size_t)n * HP + c] += fmaxf(y, 0.0f);
}

__global__ void write_out(const float* __restrict__ y3, float* __restrict__ out, int N) {
    long total = (long)N * NCLS;
    long idx   = (long)blockIdx.x * blockDim.x + threadIdx.x;
    if (idx >= total) return;
    int n = (int)(idx / NCLS);
    int c = (int)(idx - (long)n * NCLS);
    out[(size_t)n * NCLS + c] = y3[(size_t)n * 16 + c];
}

// ============================== host ==============================
extern "C" void kernel_launch(void* const* d_in, const int* in_sizes, int n_in,
                              void* d_out, int out_size, void* d_ws, size_t ws_size,
                              hipStream_t stream) {
    const float* d_h   = (const float*)d_in[0];
    const float* d_e   = (const float*)d_in[1];
    const int*   src   = (const int*)d_in[2];
    const int*   dst   = (const int*)d_in[3];
    const float* sn_n  = (const float*)d_in[4];
    const float* sn_e  = (const float*)d_in[5];
    const float* W_eh  = (const float*)d_in[6];
    const float* b_eh  = (const float*)d_in[7];
    const float* W_ee  = (const float*)d_in[8];
    const float* b_ee  = (const float*)d_in[9];
    const float* WA = (const float*)d_in[10]; const float* bA = (const float*)d_in[11];
    const float* WB = (const float*)d_in[12]; const float* bB = (const float*)d_in[13];
    const float* WC = (const float*)d_in[14]; const float* bC = (const float*)d_in[15];
    const float* WD = (const float*)d_in[16]; const float* bD = (const float*)d_in[17];
    const float* WE = (const float*)d_in[18]; const float* bE = (const float*)d_in[19];
    const float* bnh_g = (const float*)d_in[20]; const float* bnh_b = (const float*)d_in[21];
    const float* bne_g = (const float*)d_in[22]; const float* bne_b = (const float*)d_in[23];
    const float* W_m0 = (const float*)d_in[24]; const float* b_m0 = (const float*)d_in[25];
    const float* W_m1 = (const float*)d_in[26]; const float* b_m1 = (const float*)d_in[27];
    const float* W_m2 = (const float*)d_in[28]; const float* b_m2 = (const float*)d_in[29];

    const int N = in_sizes[4];   // snorm_n
    const int E = in_sizes[2];   // src

    // ---- carve workspace (256B aligned) ----
    char* p = (char*)d_ws;
    auto carve = [&](size_t bytes) -> void* {
        void* r = (void*)p;
        p += (bytes + 255) & ~(size_t)255;
        return r;
    };
    const size_t fN = (size_t)N * HP, fE = (size_t)E * HP;
    float* hbuf = (float*)carve(fN * 4);
    float* Ah   = (float*)carve(fN * 4);
    float* Bh   = (float*)carve(fN * 4);
    float* Dh   = (float*)carve(fN * 4);
    float* Eh   = (float*)carve(fN * 4);
    float* num  = (float*)carve(fN * 4);
    float* den  = (float*)carve(fN * 4);
    float* ebuf = (float*)carve(fE * 4);
    float* ehat = (float*)carve(fE * 4);
    float* y1   = (float*)carve((size_t)N * 64 * 4);
    float* y2   = (float*)carve((size_t)N * 32 * 4);
    float* y3   = (float*)carve((size_t)N * 16 * 4);
    float* sums = (float*)carve(2 * HP * 4);
    float* minv = (float*)carve(2 * HP * 4);
    unsigned short* Wehp = (unsigned short*)carve((size_t)128 * 96 * 2);
    unsigned short* Weep = (unsigned short*)carve((size_t)128 * 96 * 2);
    unsigned short* WAp  = (unsigned short*)carve((size_t)NLAY * 96 * 96 * 2);
    unsigned short* WBp  = (unsigned short*)carve((size_t)NLAY * 96 * 96 * 2);
    unsigned short* WCp  = (unsigned short*)carve((size_t)NLAY * 96 * 96 * 2);
    unsigned short* WDp  = (unsigned short*)carve((size_t)NLAY * 96 * 96 * 2);
    unsigned short* WEp  = (unsigned short*)carve((size_t)NLAY * 96 * 96 * 2);
    unsigned short* Wm0p = (unsigned short*)carve((size_t)96 * 64 * 2);
    unsigned short* Wm1p = (unsigned short*)carve((size_t)64 * 32 * 2);
    unsigned short* Wm2p = (unsigned short*)carve((size_t)32 * 16 * 2);
    (void)n_in; (void)ws_size; (void)out_size;

    auto pack = [&](const float* W, unsigned short* out, int K0, int N0, int KP, int NP) {
        int total = KP * NP;
        pack_weight<<<(total + 255) / 256, 256, 0, stream>>>(W, out, K0, N0, KP, NP);
    };
    // K / NP are compile-time shapes; dispatch to the right instantiation.
    auto gemm = [&](const float* A, const unsigned short* Bp, const float* bias, float* C,
                    int M, int K, int NP, int N0, int relu) {
        int nRT    = (M + 15) / 16;
        int blocks = (nRT + 7) / 8;                    // 8 waves / block (wave32)
        if (K == 128 && NP == 96)
            gemm_wmma<4, 6><<<blocks, 256, 0, stream>>>(A, Bp, bias, C, M, N0, relu);
        else if (K == 96 && NP == 96)
            gemm_wmma<3, 6><<<blocks, 256, 0, stream>>>(A, Bp, bias, C, M, N0, relu);
        else if (K == 96 && NP == 64)
            gemm_wmma<3, 4><<<blocks, 256, 0, stream>>>(A, Bp, bias, C, M, N0, relu);
        else if (K == 64 && NP == 32)
            gemm_wmma<2, 2><<<blocks, 256, 0, stream>>>(A, Bp, bias, C, M, N0, relu);
        else if (K == 32 && NP == 16)
            gemm_wmma<1, 1><<<blocks, 256, 0, stream>>>(A, Bp, bias, C, M, N0, relu);
    };
    auto g1d = [&](long total) { return dim3((unsigned)((total + 255) / 256)); };

    // ---- pack all weights to WMMA B-fragment bf16 layout ----
    pack(W_eh, Wehp, 128, HID, 128, HP);
    pack(W_ee, Weep, 128, HID, 128, HP);
    for (int l = 0; l < NLAY; ++l) {
        const size_t wo = (size_t)l * HID * HID, po = (size_t)l * 96 * 96;
        pack(WA + wo, WAp + po, HID, HID, HP, HP);
        pack(WB + wo, WBp + po, HID, HID, HP, HP);
        pack(WC + wo, WCp + po, HID, HID, HP, HP);
        pack(WD + wo, WDp + po, HID, HID, HP, HP);
        pack(WE + wo, WEp + po, HID, HID, HP, HP);
    }
    pack(W_m0, Wm0p, HID, 35, HP, 64);
    pack(W_m1, Wm1p, 35, 17, 64, 32);
    pack(W_m2, Wm2p, 17, NCLS, 32, 16);

    // ---- input embeddings ----
    gemm(d_h, Wehp, b_eh, hbuf, N, 128, HP, HID, 0);
    gemm(d_e, Weep, b_ee, ebuf, E, 128, HP, HID, 0);

    // ---- GatedGCN layers ----
    for (int l = 0; l < NLAY; ++l) {
        const size_t po = (size_t)l * 96 * 96;
        const size_t bo = (size_t)l * HID;
        gemm(hbuf, WAp + po, bA + bo, Ah, N, HP, HP, HID, 0);
        gemm(hbuf, WBp + po, bB + bo, Bh, N, HP, HP, HID, 0);
        gemm(hbuf, WDp + po, bD + bo, Dh, N, HP, HP, HID, 0);
        gemm(hbuf, WEp + po, bE + bo, Eh, N, HP, HP, HID, 0);
        gemm(ebuf, WCp + po, bC + bo, ehat, E, HP, HP, HID, 0);

        hipMemsetAsync(num, 0, fN * 4, stream);
        hipMemsetAsync(den, 0, fN * 4, stream);
        edge_kernel<<<g1d((long)E * HID), 256, 0, stream>>>(ehat, Dh, Eh, Bh, src, dst,
                                                            num, den, E);
        combine_h<<<g1d((long)N * HID), 256, 0, stream>>>(Ah, num, den, N);

        hipMemsetAsync(sums, 0, 2 * HP * 4, stream);
        col_stats<<<1024, 256, 0, stream>>>(Ah, sn_n, N, sums);
        finalize_stats<<<1, 128, 0, stream>>>(sums, minv, 1.0f / (float)N);
        bn_residual<<<g1d((long)N * HID), 256, 0, stream>>>(hbuf, Ah, sn_n, minv,
                                                            bnh_g + bo, bnh_b + bo, N);

        hipMemsetAsync(sums, 0, 2 * HP * 4, stream);
        col_stats<<<2048, 256, 0, stream>>>(ehat, sn_e, E, sums);
        finalize_stats<<<1, 128, 0, stream>>>(sums, minv, 1.0f / (float)E);
        bn_residual<<<g1d((long)E * HID), 256, 0, stream>>>(ebuf, ehat, sn_e, minv,
                                                            bne_g + bo, bne_b + bo, E);
    }

    // ---- MLP head ----
    gemm(hbuf, Wm0p, b_m0, y1, N, HP, 64, 35, 1);
    gemm(y1,   Wm1p, b_m1, y2, N, 64, 32, 17, 1);
    gemm(y2,   Wm2p, b_m2, y3, N, 32, 16, NCLS, 0);
    write_out<<<g1d((long)N * NCLS), 256, 0, stream>>>(y3, (float*)d_out, N);
}
